// SimpleRNN_72258529787983
// MI455X (gfx1250) — matrix operand order repaired and checked by reading, not compiled, not run on hardware
//
#include <hip/hip_runtime.h>

// ---------------------------------------------------------------------------
// SimpleRNN  (B=128, T=512, I=256, H=1024, O=256)  fp32 ref, f16 WMMA compute
//   k0: convert W_ih/W_hh/W_fc -> f16 in ws, bias_comb = b_ih+b_hh
//   k1: persistent recurrence: 8 WGs x 1024 thr, h kept in LDS across T steps
//   k2: out = h_last @ W_fc^T + b_fc  via WMMA
// d_out = [ out (128*256 f32) | hidden (128*1024 f32) ]
// ---------------------------------------------------------------------------

typedef __attribute__((ext_vector_type(16))) _Float16 v16h;
typedef __attribute__((ext_vector_type(8)))  _Float16 v8h;
typedef __attribute__((ext_vector_type(8)))  float    v8f;

#define Bn 128
#define Tn 512
#define In 256
#define Hn 1024
#define On 256

constexpr int LDX = In + 8;   // 264 halfs  (16B-aligned row stride)
constexpr int LDH = Hn + 8;   // 1032 halfs (16B-aligned row stride)

// ---- WMMA fragment loaders (contiguous k order: k = 16*hs + elem) ------------------
// A 16x32 f16, row-major src with leading dim `ld` (elements):
//   lane<16 : row=lq, elems[0..15]=K k0..k0+15
//   lane>=16: row=lq, elems[0..15]=K k0+16..k0+31
template <typename P>
__device__ __forceinline__ v16h load_a_f16(P base, int ld, int k0, int hs, int lq) {
  P p = base + lq * ld + k0 + hs * 16;
  v8h lo = *(const v8h*)(p);
  v8h hi = *(const v8h*)(p + 8);
  v16h a;
#pragma unroll
  for (int i = 0; i < 8; ++i) { a[i] = lo[i]; a[i + 8] = hi[i]; }
  return a;
}

// B 32x16 f16 where B[k][n] = W[n][k] (W row-major): per lane n=n0+lq,
// 16 contiguous K halfs starting at k0 + (lane>=16)*16.
__device__ __forceinline__ v16h load_b_f16(const _Float16* __restrict__ p) {
  v8h lo = *(const v8h*)(p);
  v8h hi = *(const v8h*)(p + 8);
  v16h b;
#pragma unroll
  for (int i = 0; i < 8; ++i) { b[i] = lo[i]; b[i + 8] = hi[i]; }
  return b;
}

__device__ __forceinline__ v8f wmma_f16(v16h a, v16h b, v8f c) {
  // D = A*B + C, emits v_wmma_f32_16x16x32_f16
  return __builtin_amdgcn_wmma_f32_16x16x32_f16(false, a, false, b, (short)0, c,
                                                false, false);
}

// tanh(x) = 1 - 2/(exp2(x*2/ln2... )+1); saturates cleanly to +-1 at large |x|
__device__ __forceinline__ float tanh_fast(float x) {
  float e = __builtin_amdgcn_exp2f(x * 2.885390081777927f);  // exp(2x)
  return 1.0f - 2.0f * __builtin_amdgcn_rcpf(e + 1.0f);
}

// ---------------------------------------------------------------------------
// k0: weight conversion + bias combine
// ---------------------------------------------------------------------------
__global__ void rnn_prep(const float* __restrict__ Wih, const float* __restrict__ Whh,
                         const float* __restrict__ Wfc, const float* __restrict__ bih,
                         const float* __restrict__ bhh,
                         _Float16* __restrict__ Wih_h, _Float16* __restrict__ Whh_h,
                         _Float16* __restrict__ Wfc_h, float* __restrict__ biasc) {
  int i = blockIdx.x * blockDim.x + threadIdx.x;
  int stride = gridDim.x * blockDim.x;
  for (int j = i; j < Hn * In; j += stride) Wih_h[j] = (_Float16)Wih[j];
  for (int j = i; j < Hn * Hn; j += stride) Whh_h[j] = (_Float16)Whh[j];
  for (int j = i; j < On * Hn; j += stride) Wfc_h[j] = (_Float16)Wfc[j];
  for (int j = i; j < Hn; j += stride) biasc[j] = bih[j] + bhh[j];
}

// ---------------------------------------------------------------------------
// k1: persistent recurrence. grid = B/16 = 8 blocks of 1024 threads.
//     Block b owns batch rows [16b, 16b+16); h slice lives in LDS as f16.
//     Wave w (of 32) owns hidden columns [32w, 32w+32) = two 16x16 N-tiles.
// ---------------------------------------------------------------------------
__global__ __launch_bounds__(1024, 1) void rnn_scan(
    const float* __restrict__ x, const float* __restrict__ hidden,
    const _Float16* __restrict__ Wih, const _Float16* __restrict__ Whh,
    const float* __restrict__ biasc, _Float16* __restrict__ hlast_h,
    float* __restrict__ hid_out) {
  __shared__ __align__(16) _Float16 s_x[16 * LDX];  // x_t tile   (~8.25 KB)
  __shared__ __align__(16) _Float16 s_h[16 * LDH];  // h state    (~32.3 KB)

  const int b0 = blockIdx.x * 16;
  const int tid = threadIdx.x;
  const int wave = tid >> 5;
  const int lane = tid & 31;
  const int hs = lane >> 4;   // half-wave select
  const int lq = lane & 15;   // row/col within tile
  const int n0 = wave * 32;   // this wave's first hidden column

  // init h from `hidden` input (f32 [1][128][1024])
  for (int j = tid; j < 16 * Hn; j += 1024) {
    int r = j >> 10, c = j & (Hn - 1);
    s_h[r * LDH + c] = (_Float16)hidden[(size_t)(b0 + r) * Hn + c];
  }

  const float bias0 = biasc[n0 + lq];
  const float bias1 = biasc[n0 + 16 + lq];
  const _Float16* wih0 = Wih + (size_t)(n0 + lq) * In + hs * 16;
  const _Float16* wih1 = Wih + (size_t)(n0 + 16 + lq) * In + hs * 16;
  const _Float16* whh0 = Whh + (size_t)(n0 + lq) * Hn + hs * 16;
  const _Float16* whh1 = Whh + (size_t)(n0 + 16 + lq) * Hn + hs * 16;

  for (int t = 0; t < Tn; ++t) {
    // stage x[:, t, :] -> s_x (f32 -> f16), 4 elems/thread
    {
      int idx = tid * 4;
      int r = idx >> 8, c = idx & (In - 1);
      const float4 v = *(const float4*)(x + ((size_t)(b0 + r) * Tn + t) * In + c);
      _Float16* dst = s_x + r * LDX + c;
      dst[0] = (_Float16)v.x; dst[1] = (_Float16)v.y;
      dst[2] = (_Float16)v.z; dst[3] = (_Float16)v.w;
    }
    __syncthreads();  // s_x staged; (t>0) h writes from prev epilogue visible

    v8f acc0 = {};
    v8f acc1 = {};
    // x_t @ W_ih^T   (K = 256)
#pragma unroll 2
    for (int k0 = 0; k0 < In; k0 += 32) {
      v16h a = load_a_f16((const _Float16*)s_x, LDX, k0, hs, lq);
      v16h bf0 = load_b_f16(wih0 + k0);
      v16h bf1 = load_b_f16(wih1 + k0);
      acc0 = wmma_f16(a, bf0, acc0);
      acc1 = wmma_f16(a, bf1, acc1);
    }
    // h_{t-1} @ W_hh^T (K = 1024)
#pragma unroll 2
    for (int k0 = 0; k0 < Hn; k0 += 32) {
      v16h a = load_a_f16((const _Float16*)s_h, LDH, k0, hs, lq);
      v16h bf0 = load_b_f16(whh0 + k0);
      v16h bf1 = load_b_f16(whh1 + k0);
      acc0 = wmma_f16(a, bf0, acc0);
      acc1 = wmma_f16(a, bf1, acc1);
    }
    __syncthreads();  // everyone done reading s_h; safe to overwrite in-place

    // epilogue: bias + tanh, write h_t back to LDS (C/D layout: M = hs*8 + r)
#pragma unroll
    for (int r = 0; r < 8; ++r) {
      int m = hs * 8 + r;
      float h0 = tanh_fast(acc0[r] + bias0);
      float h1 = tanh_fast(acc1[r] + bias1);
      s_h[m * LDH + n0 + lq] = (_Float16)h0;
      s_h[m * LDH + n0 + 16 + lq] = (_Float16)h1;
      if (t == Tn - 1) {  // final state: f32 to d_out, f16 to ws for the FC
        hid_out[(size_t)(b0 + m) * Hn + n0 + lq] = h0;
        hid_out[(size_t)(b0 + m) * Hn + n0 + 16 + lq] = h1;
        hlast_h[(size_t)(b0 + m) * Hn + n0 + lq] = (_Float16)h0;
        hlast_h[(size_t)(b0 + m) * Hn + n0 + 16 + lq] = (_Float16)h1;
      }
    }
    __syncthreads();  // h_t fully written before next step stages/reads
  }
}

// ---------------------------------------------------------------------------
// k2: out[128,256] = h_last @ W_fc^T + b_fc.  128 tiles, 1 wave each.
// ---------------------------------------------------------------------------
__global__ __launch_bounds__(256) void rnn_fc(const _Float16* __restrict__ hlh,
                                              const _Float16* __restrict__ Wfc,
                                              const float* __restrict__ bfc,
                                              float* __restrict__ out) {
  const int wave = threadIdx.x >> 5;
  const int lane = threadIdx.x & 31;
  const int tile = blockIdx.x * 8 + wave;  // 0..127
  const int m0 = (tile >> 4) * 16;         // 8 row groups
  const int n0 = (tile & 15) * 16;         // 16 col groups
  const int hs = lane >> 4, lq = lane & 15;

  v8f acc = {};
  const _Float16* wf = Wfc + (size_t)(n0 + lq) * Hn + hs * 16;
#pragma unroll 4
  for (int k0 = 0; k0 < Hn; k0 += 32) {
    v16h a = load_a_f16(hlh + (size_t)m0 * Hn, Hn, k0, hs, lq);
    v16h b = load_b_f16(wf + k0);
    acc = wmma_f16(a, b, acc);
  }
  const float bias = bfc[n0 + lq];
#pragma unroll
  for (int r = 0; r < 8; ++r) {
    int m = m0 + hs * 8 + r;
    out[(size_t)m * On + n0 + lq] = acc[r] + bias;
  }
}

// ---------------------------------------------------------------------------
extern "C" void kernel_launch(void* const* d_in, const int* in_sizes, int n_in,
                              void* d_out, int out_size, void* d_ws, size_t ws_size,
                              hipStream_t stream) {
  (void)in_sizes; (void)n_in; (void)out_size; (void)ws_size;
  const float* x   = (const float*)d_in[0];
  const float* hid = (const float*)d_in[1];
  const float* Wih = (const float*)d_in[2];
  const float* Whh = (const float*)d_in[3];
  const float* bih = (const float*)d_in[4];
  const float* bhh = (const float*)d_in[5];
  const float* Wfc = (const float*)d_in[6];
  const float* bfc = (const float*)d_in[7];

  float* out = (float*)d_out;          // [128,256]
  float* hid_out = out + Bn * On;      // [128,1024]

  // workspace layout (~3.3 MB)
  _Float16* Wih_h  = (_Float16*)d_ws;              // 1024*256
  _Float16* Whh_h  = Wih_h + Hn * In;              // 1024*1024
  _Float16* Wfc_h  = Whh_h + Hn * Hn;              // 256*1024
  float*    biasc  = (float*)(Wfc_h + On * Hn);    // 1024
  _Float16* hlast_h = (_Float16*)(biasc + Hn);     // 128*1024

  rnn_prep<<<512, 256, 0, stream>>>(Wih, Whh, Wfc, bih, bhh,
                                    Wih_h, Whh_h, Wfc_h, biasc);
  rnn_scan<<<Bn / 16, 1024, 0, stream>>>(x, hid, Wih_h, Whh_h, biasc,
                                         hlast_h, hid_out);
  rnn_fc<<<(Bn / 16) * (On / 16) / 8, 256, 0, stream>>>(hlast_h, Wfc_h, bfc, out);
}
